// NodeAttention_48756468744629
// MI455X (gfx1250) — matrix operand order repaired and checked by reading, not compiled, not run on hardware
//
#include <hip/hip_runtime.h>
#include <hip/hip_bf16.h>
#include <math.h>

typedef __attribute__((ext_vector_type(16))) _Float16 v16h;
typedef __attribute__((ext_vector_type(8)))  float    v8f;
typedef __attribute__((ext_vector_type(4)))  float    v4f;

#define N_TOK    768
#define NODE_DIM 256
#define PAIR_DIM 128
#define HEADS    8
#define DIM_HEAD 32
#define INNER    256
#define LN_EPS   1e-5f
#define NEG_MIN  (-3.4028234663852886e38f)   // jnp.finfo(float32).min

// ---------------------------------------------------------------------------
// f32 -> f16 hi/lo split (hi+lo carries ~22 mantissa bits)
// ---------------------------------------------------------------------------
__device__ inline void split16(float x, _Float16& hi, _Float16& lo) {
    _Float16 h = (_Float16)x;
    hi = h;
    lo = (_Float16)(x - (float)h);
}

// ---------------------------------------------------------------------------
// One 16x16 C += A(16x32) * B(32x16) step with f32 inputs via f16-split WMMA.
// ISA VGPR layouts (CDNA5 ISA §7.12.2):
//   A 16x32 f16 : lane<16 -> M=lane,    K = {0..7, 16..23} across elems
//                 lane>=16 -> M=lane-16, K = {8..15, 24..31}
//   B 32x16 f16 : lane<16 -> N=lane,    K = {0..7, 16..23};  lane>=16 -> K = {8..15, 24..31}
//   C 16x16 f32 : VGPR r, lane<16 -> (M=r, N=lane); lane>=16 -> (M=8+r, N=lane-16)
// ---------------------------------------------------------------------------
template <typename FA, typename FB>
__device__ inline void wmma_step_f32(v8f& c, int lane, FA&& loadA, FB&& loadB) {
    v16h ah, al, bh, bl;
    const int rowA   = lane & 15;
    const int colB   = lane & 15;
    const int hiHalf = lane >> 4;
#pragma unroll
    for (int j = 0; j < 16; ++j) {
        int kA = (hiHalf << 3) + (j & 7) + ((j >> 3) << 4);
        float a = loadA(rowA, kA);
        _Float16 h, l;
        split16(a, h, l);
        ah[j] = h; al[j] = l;

        int kB = (hiHalf << 3) + (j & 7) + ((j >> 3) << 4);
        float b = loadB(kB, colB);
        split16(b, h, l);
        bh[j] = h; bl[j] = l;
    }
    c = __builtin_amdgcn_wmma_f32_16x16x32_f16(false, ah, false, bh, (short)0, c, false, false);
    c = __builtin_amdgcn_wmma_f32_16x16x32_f16(false, ah, false, bl, (short)0, c, false, false);
    c = __builtin_amdgcn_wmma_f32_16x16x32_f16(false, al, false, bh, (short)0, c, false, false);
}

// ---------------------------------------------------------------------------
// 1. LayerNorm of node_feats: one 256-thread block per row of 256 elements
// ---------------------------------------------------------------------------
__global__ void ln_node_kernel(const float* __restrict__ x,
                               const float* __restrict__ w,
                               const float* __restrict__ b,
                               float* __restrict__ y) {
    int row = blockIdx.x;
    int t   = threadIdx.x;               // 256 threads
    float v = x[row * NODE_DIM + t];
    float s = v, sq = v * v;
    for (int off = 16; off; off >>= 1) {
        s  += __shfl_xor(s,  off, 32);
        sq += __shfl_xor(sq, off, 32);
    }
    __shared__ float ssum[8], ssq[8];
    int wave = t >> 5, lane = t & 31;
    if (lane == 0) { ssum[wave] = s; ssq[wave] = sq; }
    __syncthreads();
    if (t == 0) {
        float S = 0.f, Q = 0.f;
        for (int i = 0; i < 8; ++i) { S += ssum[i]; Q += ssq[i]; }
        float mu  = S / (float)NODE_DIM;
        float var = Q / (float)NODE_DIM - mu * mu;
        if (var < 0.f) var = 0.f;
        ssum[0] = mu;
        ssq[0]  = rsqrtf(var + LN_EPS);
    }
    __syncthreads();
    float mu = ssum[0], rs = ssq[0];
    y[row * NODE_DIM + t] = (v - mu) * rs * w[t] + b[t];
}

// ---------------------------------------------------------------------------
// 2. Fused qkv + gate GEMM: xln(768x256) @ [w_qkv(256x768) | w_g(256x256)]
//    One wave per 16x16 tile; 3072 waves total.
// ---------------------------------------------------------------------------
__global__ void qkvg_kernel(const float* __restrict__ xln,
                            const float* __restrict__ w_qkv,
                            const float* __restrict__ w_g,
                            const float* __restrict__ b_g,
                            float* __restrict__ q, float* __restrict__ kb,
                            float* __restrict__ vb, float* __restrict__ g) {
    int wave = (blockIdx.x * blockDim.x + threadIdx.x) >> 5;
    int lane = threadIdx.x & 31;
    int tileM = wave >> 6;      // 48 tiles of M
    int tileN = wave & 63;      // 64 tiles of N (1024 cols)
    int n0 = tileN * 16;
    v8f c = {};
    for (int kk = 0; kk < NODE_DIM; kk += 32) {
        wmma_step_f32(c, lane,
            [&](int r, int kx) {
                return xln[(tileM * 16 + r) * NODE_DIM + kk + kx];
            },
            [&](int kx, int cn) {
                int ncol = n0 + cn;
                return (ncol < 768) ? w_qkv[(kk + kx) * 768 + ncol]
                                    : w_g[(kk + kx) * 256 + (ncol - 768)];
            });
    }
    int col = n0 + (lane & 15);
#pragma unroll
    for (int r = 0; r < 8; ++r) {
        int row = tileM * 16 + r + ((lane >> 4) << 3);
        float val = c[r];
        if      (col < 256) q [row * 256 +  col]        = val;
        else if (col < 512) kb[row * 256 + (col - 256)] = val;
        else if (col < 768) vb[row * 256 + (col - 512)] = val;
        else                g [row * 256 + (col - 768)] = val + b_g[col - 768];
    }
}

// ---------------------------------------------------------------------------
// 3. sim[h,i,j] = mask ? (q_h . k_h) * scale : -FLT_MAX.  One wave per
//    (head, 16x16 tile); K = DIM_HEAD = 32 -> single WMMA step (x3 split).
// ---------------------------------------------------------------------------
__global__ void qk_kernel(const float* __restrict__ q,
                          const float* __restrict__ kmat,
                          const int* __restrict__ mask,
                          float* __restrict__ sim) {
    int wave = (blockIdx.x * blockDim.x + threadIdx.x) >> 5;
    int lane = threadIdx.x & 31;
    int h   = wave / (48 * 48);
    int rem = wave % (48 * 48);
    int ti = rem / 48, tj = rem % 48;
    v8f c = {};
    wmma_step_f32(c, lane,
        [&](int r, int kx)  { return q   [(ti * 16 + r ) * INNER + h * 32 + kx]; },
        [&](int kx, int cn) { return kmat[(tj * 16 + cn) * INNER + h * 32 + kx]; });
    const float scale = 0.17677669529663687f;   // 32^-0.5
    int j = tj * 16 + (lane & 15);
#pragma unroll
    for (int r = 0; r < 8; ++r) {
        int i = ti * 16 + r + ((lane >> 4) << 3);
        float s = c[r] * scale;
        int m = mask[i * N_TOK + j];
        sim[((size_t)h * N_TOK + i) * N_TOK + j] = m ? s : NEG_MIN;
    }
}

// ---------------------------------------------------------------------------
// 4. Pair LN + bias einsum, added into sim.  One wave per (i,j): reads the
//    512-byte pair row exactly once (non-temporal, single-use 302 MB tensor).
// ---------------------------------------------------------------------------
__global__ void pair_bias_kernel(const float* __restrict__ p,
                                 const float* __restrict__ wln,
                                 const float* __restrict__ bln,
                                 const float* __restrict__ w_bias,
                                 float* __restrict__ sim) {
    int wave = (blockIdx.x * blockDim.x + threadIdx.x) >> 5;
    int lane = threadIdx.x & 31;
    int i = wave / N_TOK, j = wave % N_TOK;
    const v4f* prow = (const v4f*)(p + ((size_t)i * N_TOK + j) * PAIR_DIM);
    v4f pv = __builtin_nontemporal_load(&prow[lane]);   // 16B/lane, fully coalesced
    float s  = pv[0] + pv[1] + pv[2] + pv[3];
    float sq = pv[0] * pv[0] + pv[1] * pv[1] + pv[2] * pv[2] + pv[3] * pv[3];
    for (int off = 16; off; off >>= 1) {
        s  += __shfl_xor(s,  off, 32);
        sq += __shfl_xor(sq, off, 32);
    }
    float mu  = s / (float)PAIR_DIM;
    float var = sq / (float)PAIR_DIM - mu * mu;
    if (var < 0.f) var = 0.f;
    float rs = rsqrtf(var + LN_EPS);

    float acc[HEADS];
#pragma unroll
    for (int h = 0; h < HEADS; ++h) acc[h] = 0.f;
    int c0 = lane * 4;
#pragma unroll
    for (int cc = 0; cc < 4; ++cc) {
        int cidx = c0 + cc;
        float pn = (pv[cc] - mu) * rs * wln[cidx] + bln[cidx];
#pragma unroll
        for (int h = 0; h < HEADS; ++h) acc[h] += pn * w_bias[cidx * HEADS + h];
    }
#pragma unroll
    for (int h = 0; h < HEADS; ++h)
        for (int off = 16; off; off >>= 1) acc[h] += __shfl_xor(acc[h], off, 32);
    if (lane == 0) {
#pragma unroll
        for (int h = 0; h < HEADS; ++h) {
            size_t idx = ((size_t)h * N_TOK + i) * N_TOK + j;
            sim[idx] += acc[h];
        }
    }
}

// ---------------------------------------------------------------------------
// 5. Row softmax in-place: one wave per (h,i) row of 768, 24 elems/lane
// ---------------------------------------------------------------------------
__global__ void softmax_kernel(float* __restrict__ sim) {
    int wave = (blockIdx.x * blockDim.x + threadIdx.x) >> 5;   // 8*768 rows
    int lane = threadIdx.x & 31;
    float* row = sim + (size_t)wave * N_TOK;
    float vals[24];
    float mx = -INFINITY;
#pragma unroll
    for (int t = 0; t < 24; ++t) {
        vals[t] = row[lane + t * 32];
        mx = fmaxf(mx, vals[t]);
    }
    for (int off = 16; off; off >>= 1) mx = fmaxf(mx, __shfl_xor(mx, off, 32));
    float sum = 0.f;
#pragma unroll
    for (int t = 0; t < 24; ++t) {
        vals[t] = expf(vals[t] - mx);
        sum += vals[t];
    }
    for (int off = 16; off; off >>= 1) sum += __shfl_xor(sum, off, 32);
    float inv = 1.f / sum;
#pragma unroll
    for (int t = 0; t < 24; ++t) row[lane + t * 32] = vals[t] * inv;
}

// ---------------------------------------------------------------------------
// 6. O = attn @ V per head, fused sigmoid gate.  One wave per 16x16 tile.
// ---------------------------------------------------------------------------
__global__ void av_gate_kernel(const float* __restrict__ attn,
                               const float* __restrict__ vmat,
                               const float* __restrict__ g,
                               float* __restrict__ gated) {
    int wave = (blockIdx.x * blockDim.x + threadIdx.x) >> 5;
    int lane = threadIdx.x & 31;
    int h   = wave / 96;        // 48 ti * 2 tn tiles per head
    int rem = wave % 96;
    int ti = rem / 2, tn = rem % 2;
    const float* arow = attn + (size_t)h * N_TOK * N_TOK;
    v8f c = {};
    for (int kk = 0; kk < N_TOK; kk += 32) {
        wmma_step_f32(c, lane,
            [&](int r, int kx)  { return arow[(ti * 16 + r) * N_TOK + kk + kx]; },
            [&](int kx, int cn) { return vmat[(kk + kx) * INNER + h * 32 + tn * 16 + cn]; });
    }
    int colg = h * 32 + tn * 16 + (lane & 15);
#pragma unroll
    for (int r = 0; r < 8; ++r) {
        int i = ti * 16 + r + ((lane >> 4) << 3);
        float gv = g[i * INNER + colg];
        float sg = 1.f / (1.f + expf(-gv));
        gated[i * INNER + colg] = c[r] * sg;
    }
}

// ---------------------------------------------------------------------------
// 7. out = gated(768x256) @ w_out(256x256) + b_out
// ---------------------------------------------------------------------------
__global__ void out_kernel(const float* __restrict__ gated,
                           const float* __restrict__ w_out,
                           const float* __restrict__ b_out,
                           float* __restrict__ out) {
    int wave = (blockIdx.x * blockDim.x + threadIdx.x) >> 5;
    int lane = threadIdx.x & 31;
    int ti = wave / 16, tn = wave % 16;
    v8f c = {};
    for (int kk = 0; kk < INNER; kk += 32) {
        wmma_step_f32(c, lane,
            [&](int r, int kx)  { return gated[(ti * 16 + r) * INNER + kk + kx]; },
            [&](int kx, int cn) { return w_out[(kk + kx) * NODE_DIM + tn * 16 + cn]; });
    }
    int n = tn * 16 + (lane & 15);
#pragma unroll
    for (int r = 0; r < 8; ++r) {
        int i = ti * 16 + r + ((lane >> 4) << 3);
        out[i * NODE_DIM + n] = c[r] + b_out[n];
    }
}

// ---------------------------------------------------------------------------
extern "C" void kernel_launch(void* const* d_in, const int* in_sizes, int n_in,
                              void* d_out, int out_size, void* d_ws, size_t ws_size,
                              hipStream_t stream) {
    const float* node_feats = (const float*)d_in[0];
    const float* pair_feats = (const float*)d_in[1];
    const int*   mask       = (const int*)d_in[2];
    const float* ln_node_w  = (const float*)d_in[3];
    const float* ln_node_b  = (const float*)d_in[4];
    const float* ln_pair_w  = (const float*)d_in[5];
    const float* ln_pair_b  = (const float*)d_in[6];
    const float* w_qkv      = (const float*)d_in[7];
    const float* w_g        = (const float*)d_in[8];
    const float* b_g        = (const float*)d_in[9];
    const float* w_bias     = (const float*)d_in[10];
    const float* w_out      = (const float*)d_in[11];
    const float* b_out      = (const float*)d_in[12];
    float* out = (float*)d_out;

    // workspace layout (floats)
    float* ws    = (float*)d_ws;
    const size_t NM = (size_t)N_TOK * NODE_DIM;   // 196608
    float* xln   = ws;
    float* q     = xln   + NM;
    float* kb    = q     + NM;
    float* vb    = kb    + NM;
    float* g     = vb    + NM;
    float* gated = g     + NM;
    float* sim   = gated + NM;                    // 8*768*768 = 4,718,592 floats

    // 1. node layernorm
    ln_node_kernel<<<N_TOK, 256, 0, stream>>>(node_feats, ln_node_w, ln_node_b, xln);
    // 2. qkv + gate GEMM  (3072 waves)
    qkvg_kernel<<<768, 128, 0, stream>>>(xln, w_qkv, w_g, b_g, q, kb, vb, g);
    // 3. masked scaled QK^T  (18432 waves)
    qk_kernel<<<4608, 128, 0, stream>>>(q, kb, mask, sim);
    // 4. pair LN + bias einsum, added into sim  (589824 waves — the 302 MB read)
    pair_bias_kernel<<<73728, 256, 0, stream>>>(pair_feats, ln_pair_w, ln_pair_b,
                                                w_bias, sim);
    // 5. softmax  (6144 rows)
    softmax_kernel<<<768, 256, 0, stream>>>(sim);
    // 6. attn @ V with sigmoid gate  (768 waves)
    av_gate_kernel<<<192, 128, 0, stream>>>(sim, vb, g, gated);
    // 7. output projection  (768 waves)
    out_kernel<<<192, 128, 0, stream>>>(gated, w_out, b_out, out);
}
